// DeepSeekMoE_10754598109817
// MI455X (gfx1250) — matrix operand order repaired and checked by reading, not compiled, not run on hardware
//
#include <hip/hip_runtime.h>
#include <hip/hip_bf16.h>

// ---------------------------------------------------------------------------
// DeepSeek MoE for gfx1250 (MI455X): sparse top-2 routed expert FFN on
// v_wmma_f32_16x16x32_bf16. Expert weights are pre-converted once per launch
// into a bf16 workspace in contiguous-k B-fragment order (lane-major 32x16 tiles), so
// the GEMM inner loop feeds the matrix pipe with two global_load_b128 per
// fragment (immediate-offset addressed) and no conversion VALU. Prepped
// weights (50 MB) are L2-resident (192 MB L2). Falls back to on-the-fly
// fp32->bf16 B loads if the workspace is too small.
// ---------------------------------------------------------------------------

#define D_MODEL 1024
#define HIDDEN  512
#define NEXP    16
#define MT      32    // token-tile rows per workgroup in expert FFN

typedef __attribute__((ext_vector_type(16))) __bf16 v16bf;
typedef __attribute__((ext_vector_type(8)))  float  v8f;

__device__ __forceinline__ unsigned short f2bf(float f) {
    // round-to-nearest-even fp32 -> bf16
    unsigned u = __float_as_uint(f);
    unsigned r = u + 0x7FFFu + ((u >> 16) & 1u);
    return (unsigned short)(r >> 16);
}

__device__ __forceinline__ v8f wmma_bf16(v16bf a, v16bf b, v8f c) {
    // D = A(16x32 bf16) * B(32x16 bf16) + C(16x16 f32)
    return __builtin_amdgcn_wmma_f32_16x16x32_bf16(
        /*neg_a=*/false, a, /*neg_b=*/false, b,
        /*c_mod=*/(short)0, c, /*reuse_a=*/false, /*reuse_b=*/false);
}

// A fragment (16x32 bf16) from LDS, contiguous-k layout:
// lanes 0-15: M=lane,    elems 0-15 -> K=0..15
// lanes16-31: M=lane-16, elems 0-15 -> K=16..31
__device__ __forceinline__ v16bf load_a_frag(const unsigned short* base, int ld,
                                             int mbase, int k0, int lane) {
    int m  = mbase + (lane & 15);
    int kb = k0 + ((lane < 16) ? 0 : 16);
    const unsigned short* row = base + m * ld + kb;
    union { v16bf v; unsigned short s[16]; } f;
#pragma unroll
    for (int j = 0; j < 8; ++j) {
        f.s[j]     = row[j];
        f.s[8 + j] = row[8 + j];
    }
    return f.v;
}

// B fragment from prepped bf16 weights at a precomputed lane pointer:
// two 16-byte vector loads; caller supplies base + compile-time tile offset
// so the offsets fold into the global_load_b128 immediate field.
__device__ __forceinline__ v16bf load_b_ptr(const unsigned short* __restrict__ p) {
    union { v16bf v; uint4 q[2]; } f;
    f.q[0] = ((const uint4*)p)[0];     // global_load_b128
    f.q[1] = ((const uint4*)p)[1];     // global_load_b128 (+16)
    return f.v;
}

// Fallback B fragment (32x16 bf16) from row-major fp32 W[ld cols]:
// lanes 0-15: K=krow+0..15 ; lanes 16-31: K=krow+16..31 (koff folded by caller)
__device__ __forceinline__ v16bf load_b_f32(const float* __restrict__ W, int ld,
                                            int krow, int n) {
    const float* col = W + (size_t)krow * ld + n;
    union { v16bf v; unsigned short s[16]; } f;
#pragma unroll
    for (int j = 0; j < 16; ++j)
        f.s[j] = f2bf(col[(size_t)j * ld]);
    return f.v;
}

// ---------------------------------------------------------------------------
// 0) Weight prep: fp32 [K,N] row-major -> bf16 fragment-tiled.
//    Element (k,n): kc=k/32, kl=k%32, nt=n/16, nl=n%16,
//    lane = (kl<16) ? nl : 16+nl, j = kl&15,
//    dst[ (kc*(N/16)+nt)*512 + lane*16 + j ].
// ---------------------------------------------------------------------------
__global__ __launch_bounds__(256) void moe_prep_kernel(
    const float* __restrict__ src, unsigned short* __restrict__ dst,
    int logN, size_t total)
{
    size_t idx = (size_t)blockIdx.x * 256 + threadIdx.x;
    if (idx >= total) return;
    const int N  = 1 << logN;
    const int k  = (int)(idx >> logN);
    const int n  = (int)(idx & (N - 1));
    const int kc = k >> 5, kl = k & 31;
    const int nt = n >> 4, nl = n & 15;
    const int lane = (kl < 16) ? nl : (16 + nl);
    const int j    = kl & 15;
    const size_t o = (((size_t)(kc * (N >> 4) + nt)) << 9) + lane * 16 + j;
    dst[o] = f2bf(src[idx]);
}

// ---------------------------------------------------------------------------
// 1) Gating: one wave per token.
// ---------------------------------------------------------------------------
__global__ __launch_bounds__(256) void moe_gate_kernel(
    const float* __restrict__ x, const float* __restrict__ gw,
    const float* __restrict__ gb, int* __restrict__ cnt,
    int* __restrict__ eidx, float* __restrict__ ew, int T)
{
    const int lane = threadIdx.x & 31;
    const int t = blockIdx.x * 8 + (threadIdx.x >> 5);
    if (t >= T) return;

    float acc[NEXP];
#pragma unroll
    for (int e = 0; e < NEXP; ++e) acc[e] = 0.0f;

    const float* xr = x + (size_t)t * D_MODEL;
#pragma unroll 4
    for (int i = 0; i < 32; ++i) {
        int d = i * 32 + lane;
        float xv = xr[d];
        const float4* g4 = reinterpret_cast<const float4*>(gw + d * NEXP);
#pragma unroll
        for (int q = 0; q < 4; ++q) {
            float4 g = g4[q];
            acc[q * 4 + 0] += xv * g.x;
            acc[q * 4 + 1] += xv * g.y;
            acc[q * 4 + 2] += xv * g.z;
            acc[q * 4 + 3] += xv * g.w;
        }
    }
#pragma unroll
    for (int e = 0; e < NEXP; ++e)
#pragma unroll
        for (int off = 16; off > 0; off >>= 1)
            acc[e] += __shfl_xor(acc[e], off, 32);

    float mx = -1e30f;
#pragma unroll
    for (int e = 0; e < NEXP; ++e) { acc[e] += gb[e]; mx = fmaxf(mx, acc[e]); }
    float s = 0.0f;
#pragma unroll
    for (int e = 0; e < NEXP; ++e) { acc[e] = __expf(acc[e] - mx); s += acc[e]; }
    float inv = 1.0f / s;

    int i1 = 0; float p1 = acc[0];
#pragma unroll
    for (int e = 1; e < NEXP; ++e) if (acc[e] > p1) { p1 = acc[e]; i1 = e; }
    int i2 = -1; float p2 = -1e30f;
#pragma unroll
    for (int e = 0; e < NEXP; ++e)
        if (e != i1 && acc[e] > p2) { p2 = acc[e]; i2 = e; }

    p1 *= inv; p2 *= inv;
    float ssum = p1 + p2 + 1e-9f;

    if (lane == 0) {
        eidx[2 * t]     = i1;
        eidx[2 * t + 1] = i2;
        ew[2 * t]       = p1 / ssum;
        ew[2 * t + 1]   = p2 / ssum;
        atomicAdd(&cnt[i1], 1);
        atomicAdd(&cnt[i2], 1);
    }
}

// ---------------------------------------------------------------------------
// 2) Prefix sum over expert counts + aux loss.
// ---------------------------------------------------------------------------
__global__ void moe_finalize_kernel(const int* __restrict__ cnt,
                                    int* __restrict__ cursor,
                                    int* __restrict__ offs,
                                    float* __restrict__ out_aux, int T)
{
    if (threadIdx.x == 0 && blockIdx.x == 0) {
        int run = 0;
        float pen = 0.0f;
        float cap = 1.25f * (float)(T * 2) / (float)NEXP;
        for (int e = 0; e < NEXP; ++e) {
            offs[e] = run;
            cursor[e] = run;
            run += cnt[e];
            float ov = (float)cnt[e] - cap;
            if (ov > 0.0f) pen += ov;
        }
        out_aux[0] = 0.01f * pen / (float)NEXP / (float)T;
    }
}

// ---------------------------------------------------------------------------
// 3) Scatter: build contiguous per-expert (token, weight) lists.
// ---------------------------------------------------------------------------
__global__ __launch_bounds__(256) void moe_scatter_kernel(
    const int* __restrict__ eidx, const float* __restrict__ ew,
    int* __restrict__ cursor, int* __restrict__ rowtok,
    float* __restrict__ roww, int T)
{
    int t = blockIdx.x * 256 + threadIdx.x;
    if (t >= T) return;
#pragma unroll
    for (int k = 0; k < 2; ++k) {
        int e = eidx[2 * t + k];
        int pos = atomicAdd(&cursor[e], 1);
        rowtok[pos] = t;
        roww[pos] = ew[2 * t + k];
    }
}

// ---------------------------------------------------------------------------
// 4) Fused expert FFN per (expert, 32-token tile).
//    8 waves: phase A each wave owns 64 of 512 hidden cols;
//             phase B each wave owns 128 of 1024 output cols.
// ---------------------------------------------------------------------------
template <bool PREPPED>
__global__ __launch_bounds__(256) void moe_ffn_kernel(
    const float* __restrict__ x,
    const float* __restrict__ gp_w, const float* __restrict__ gp_b,
    const float* __restrict__ up_w, const float* __restrict__ up_b,
    const float* __restrict__ dp_w, const float* __restrict__ dp_b,
    const unsigned short* __restrict__ gpP,
    const unsigned short* __restrict__ upP,
    const unsigned short* __restrict__ dpP,
    const int* __restrict__ cnt, const int* __restrict__ offs,
    const int* __restrict__ rowtok, const float* __restrict__ roww,
    float* __restrict__ out)
{
    const int e = blockIdx.y;
    const int n_e = cnt[e];
    const int row0 = blockIdx.x * MT;
    if (row0 >= n_e) return;   // uniform over block -> EXEC stays all-ones
    const int base = offs[e];

    const int tid  = threadIdx.x;
    const int lane = tid & 31;
    const int wv   = tid >> 5;

    __shared__ int            sTok[MT];
    __shared__ float          sWt[MT];
    __shared__ unsigned short sA[MT][40];          // x k-chunk, bf16, padded
    __shared__ unsigned short sH[MT][HIDDEN + 8];  // h, bf16

    if (tid < MT) {
        int r = row0 + tid;
        bool v = r < n_e;
        sTok[tid] = v ? rowtok[base + r] : 0;      // dup row 0 with weight 0
        sWt[tid]  = v ? roww[base + r] : 0.0f;
    }
    __syncthreads();

    const float* gW = gp_w + (size_t)e * D_MODEL * HIDDEN;
    const float* uW = up_w + (size_t)e * D_MODEL * HIDDEN;

    const v8f vz = {0.f, 0.f, 0.f, 0.f, 0.f, 0.f, 0.f, 0.f};
    v8f accg[2][4], accu[2][4];
#pragma unroll
    for (int mi = 0; mi < 2; ++mi)
#pragma unroll
        for (int ns = 0; ns < 4; ++ns) { accg[mi][ns] = vz; accu[mi][ns] = vz; }

    const int ncol0 = wv * 64;
    const int nlo   = lane & 15;
    const int koff  = (lane < 16) ? 0 : 16;
    const int mhi   = (lane < 16) ? 0 : 8;

    // Hoisted gather pointers: this thread always fills sA[mg+{0,8,16,24}][kg].
    const int kg = tid & 31;
    const int mg = tid >> 5;
    const float* xp0 = x + (size_t)sTok[mg +  0] * D_MODEL + kg;
    const float* xp1 = x + (size_t)sTok[mg +  8] * D_MODEL + kg;
    const float* xp2 = x + (size_t)sTok[mg + 16] * D_MODEL + kg;
    const float* xp3 = x + (size_t)sTok[mg + 24] * D_MODEL + kg;

    // Hoisted prepped-B base pointers (advance by constant stride per k-chunk;
    // per-ns tiles addressed with compile-time immediate offsets).
    const unsigned short* pG =
        gpP + (((size_t)(e * 32) * 32 + wv * 4) << 9) + lane * 16;   // kcA0=e*32, Ntiles=32
    const unsigned short* pU =
        upP + (((size_t)(e * 32) * 32 + wv * 4) << 9) + lane * 16;

    // ---- phase A: gate & up projections over K = D_MODEL ----
    for (int k0 = 0; k0 < D_MODEL; k0 += 32) {
        __syncthreads();
        sA[mg +  0][kg] = f2bf(xp0[k0]);
        sA[mg +  8][kg] = f2bf(xp1[k0]);
        sA[mg + 16][kg] = f2bf(xp2[k0]);
        sA[mg + 24][kg] = f2bf(xp3[k0]);
        __syncthreads();

        v16bf af0 = load_a_frag(&sA[0][0], 40, 0, 0, lane);
        v16bf af1 = load_a_frag(&sA[0][0], 40, 16, 0, lane);
#pragma unroll
        for (int ns = 0; ns < 4; ++ns) {
            v16bf bg, bu;
            if constexpr (PREPPED) {
                bg = load_b_ptr(pG + ns * 512);
                bu = load_b_ptr(pU + ns * 512);
            } else {
                int n = ncol0 + ns * 16 + nlo;
                bg = load_b_f32(gW, HIDDEN, k0 + koff, n);
                bu = load_b_f32(uW, HIDDEN, k0 + koff, n);
            }
            accg[0][ns] = wmma_bf16(af0, bg, accg[0][ns]);
            accg[1][ns] = wmma_bf16(af1, bg, accg[1][ns]);
            accu[0][ns] = wmma_bf16(af0, bu, accu[0][ns]);
            accu[1][ns] = wmma_bf16(af1, bu, accu[1][ns]);
        }
        pG += 32 * 512;    // next k-chunk row of tiles
        pU += 32 * 512;
    }

    // epilogue: h = gate * silu(up)  -> LDS bf16
#pragma unroll
    for (int mi = 0; mi < 2; ++mi)
#pragma unroll
        for (int ns = 0; ns < 4; ++ns) {
            int n = ncol0 + ns * 16 + nlo;
            float gbv = gp_b[e * HIDDEN + n];
            float ubv = up_b[e * HIDDEN + n];
#pragma unroll
            for (int r = 0; r < 8; ++r) {
                float g = accg[mi][ns][r] + gbv;
                float u = accu[mi][ns][r] + ubv;
                float su = u / (1.0f + __expf(-u));    // silu
                sH[mi * 16 + mhi + r][n] = f2bf(g * su);
            }
        }
    __syncthreads();

    // ---- phase B: down projection over K = HIDDEN ----
    const float* dW = dp_w + (size_t)e * HIDDEN * D_MODEL;
    v8f accy[2][8];
#pragma unroll
    for (int mi = 0; mi < 2; ++mi)
#pragma unroll
        for (int ns = 0; ns < 8; ++ns) accy[mi][ns] = vz;

    const int nb0 = wv * 128;
    const unsigned short* pD =
        dpP + (((size_t)(e * 16) * 64 + wv * 8) << 9) + lane * 16;   // kcB0=e*16, Ntiles=64

    for (int k0 = 0; k0 < HIDDEN; k0 += 32) {
        v16bf af0 = load_a_frag(&sH[0][0], HIDDEN + 8, 0, k0, lane);
        v16bf af1 = load_a_frag(&sH[0][0], HIDDEN + 8, 16, k0, lane);
#pragma unroll
        for (int ns = 0; ns < 8; ++ns) {
            v16bf bd;
            if constexpr (PREPPED) {
                bd = load_b_ptr(pD + ns * 512);
            } else {
                int n = nb0 + ns * 16 + nlo;
                bd = load_b_f32(dW, D_MODEL, k0 + koff, n);
            }
            accy[0][ns] = wmma_bf16(af0, bd, accy[0][ns]);
            accy[1][ns] = wmma_bf16(af1, bd, accy[1][ns]);
        }
        pD += 64 * 512;
    }

    // combine: out[tok] += weight * (y + dp_b)
#pragma unroll
    for (int mi = 0; mi < 2; ++mi)
#pragma unroll
        for (int r = 0; r < 8; ++r) {
            int m = mi * 16 + mhi + r;
            int tok = sTok[m];
            float wt = sWt[m];
#pragma unroll
            for (int ns = 0; ns < 8; ++ns) {
                int n = nb0 + ns * 16 + nlo;
                float val = accy[mi][ns][r] + dp_b[e * D_MODEL + n];
                atomicAdd(&out[(size_t)tok * D_MODEL + n], wt * val);
            }
        }
}

// ---------------------------------------------------------------------------
// Host launch
// ---------------------------------------------------------------------------
extern "C" void kernel_launch(void* const* d_in, const int* in_sizes, int n_in,
                              void* d_out, int out_size, void* d_ws, size_t ws_size,
                              hipStream_t stream)
{
    const float* x      = (const float*)d_in[0];
    const float* gate_w = (const float*)d_in[1];
    const float* gate_b = (const float*)d_in[2];
    const float* gp_w   = (const float*)d_in[3];
    const float* gp_b   = (const float*)d_in[4];
    const float* up_w   = (const float*)d_in[5];
    const float* up_b   = (const float*)d_in[6];
    const float* dp_w   = (const float*)d_in[7];
    const float* dp_b   = (const float*)d_in[8];
    float* out = (float*)d_out;

    const int T = in_sizes[0] / D_MODEL;   // 16384 tokens

    // workspace layout: routing data, then (optionally) prepped bf16 weights
    int*   cnt    = (int*)d_ws;               // [16]
    int*   cursor = cnt + NEXP;               // [16]
    int*   offs   = cursor + NEXP;            // [16]
    int*   eidx   = offs + NEXP;              // [2T]
    int*   rowtok = eidx + 2 * T;             // [2T]
    float* ew     = (float*)(rowtok + 2 * T); // [2T]
    float* roww   = ew + 2 * T;               // [2T]

    size_t routeBytes = (size_t)(3 * NEXP + 8 * T) * 4;
    routeBytes = (routeBytes + 255) & ~(size_t)255;

    const size_t wElems = (size_t)NEXP * D_MODEL * HIDDEN;  // per matrix set
    const size_t needBytes = routeBytes + 3 * wElems * sizeof(unsigned short);
    const bool prepped = (ws_size >= needBytes);

    unsigned short* gpP = (unsigned short*)((char*)d_ws + routeBytes);
    unsigned short* upP = gpP + wElems;
    unsigned short* dpP = upP + wElems;

    hipMemsetAsync(d_out, 0, (size_t)T * D_MODEL * sizeof(float), stream);
    hipMemsetAsync(cnt, 0, NEXP * sizeof(int), stream);

    if (prepped) {
        int nb = (int)((wElems + 255) / 256);
        // gp/up: [E*D_MODEL, HIDDEN] (logN=9); dp: [E*HIDDEN, D_MODEL] (logN=10)
        moe_prep_kernel<<<nb, 256, 0, stream>>>(gp_w, gpP, 9,  wElems);
        moe_prep_kernel<<<nb, 256, 0, stream>>>(up_w, upP, 9,  wElems);
        moe_prep_kernel<<<nb, 256, 0, stream>>>(dp_w, dpP, 10, wElems);
    }

    moe_gate_kernel<<<(T + 7) / 8, 256, 0, stream>>>(x, gate_w, gate_b,
                                                     cnt, eidx, ew, T);
    moe_finalize_kernel<<<1, 32, 0, stream>>>(cnt, cursor, offs,
                                              out + (size_t)T * D_MODEL, T);
    moe_scatter_kernel<<<(T + 255) / 256, 256, 0, stream>>>(eidx, ew, cursor,
                                                            rowtok, roww, T);
    dim3 grid((T + MT - 1) / MT, NEXP);
    if (prepped) {
        moe_ffn_kernel<true><<<grid, 256, 0, stream>>>(
            x, gp_w, gp_b, up_w, up_b, dp_w, dp_b, gpP, upP, dpP,
            cnt, offs, rowtok, roww, out);
    } else {
        moe_ffn_kernel<false><<<grid, 256, 0, stream>>>(
            x, gp_w, gp_b, up_w, up_b, dp_w, dp_b, gpP, upP, dpP,
            cnt, offs, rowtok, roww, out);
    }
}
